// CombinedRepeatCausalLinear_33011118637629
// MI455X (gfx1250) — matrix operand order repaired and checked by reading, not compiled, not run on hardware
//
#include <hip/hip_runtime.h>
#include <stdint.h>

// Problem geometry (from reference)
#define S_DIM 2048            // sequence length S == N == K
#define M_TOT 8192            // B*EMB = 8*1024 rows of X
#define BM 128
#define BN 128

typedef __attribute__((ext_vector_type(16))) __bf16 v16bf;
typedef __attribute__((ext_vector_type(8)))  float  v8f;

union FragU { uint4 q[2]; v16bf v; };

static __device__ __forceinline__ unsigned short f32_to_bf16(float f) {
    union { float f; uint32_t u; } c; c.f = f;
    uint32_t u = c.u;
    uint32_t r = u + 0x7FFFu + ((u >> 16) & 1u);   // round-to-nearest-even
    return (unsigned short)(r >> 16);
}

// ---------------------------------------------------------------------------
// Kernel 1: materialize Wt[t][s] = (s<=t) ? w0[s]*d0^(t-s) + w1[t]*d1^(t-s) : 0
// stored bf16, N-major (row t, col s) so the GEMM B-operand is K-contiguous.
// ---------------------------------------------------------------------------
__global__ __launch_bounds__(256) void crcl_build_wt(
    const float* __restrict__ weight,      // (2, S)
    const float* __restrict__ decay,       // (2, 1)
    unsigned short* __restrict__ wt)       // (S, S) bf16
{
    const int t  = blockIdx.x;
    const int s0 = threadIdx.x * 8;

    const float d0   = fminf(fmaxf(decay[0], 0.9f), 1.0f);
    const float d1   = fminf(fmaxf(decay[1], 0.9f), 1.0f);
    const float l2d0 = __builtin_amdgcn_logf(d0);   // v_log_f32 == log2
    const float l2d1 = __builtin_amdgcn_logf(d1);
    const float w1t  = weight[S_DIM + t];

    unsigned short o[8];
#pragma unroll
    for (int i = 0; i < 8; ++i) {
        const int s = s0 + i;
        float v = 0.0f;
        if (s <= t) {
            const float pw = (float)(t - s);          // (j-i)/DC, DC == 1.0
            v = weight[s] * __builtin_amdgcn_exp2f(pw * l2d0)
              + w1t      * __builtin_amdgcn_exp2f(pw * l2d1);
        }
        o[i] = f32_to_bf16(v);
    }
    *(uint4*)(&wt[(size_t)t * S_DIM + s0]) = *(const uint4*)o;
}

// ---------------------------------------------------------------------------
// Kernel 1b: one-shot X fp32 -> bf16 (removes conversion from GEMM hot loop,
// halves A-operand bytes; whole bf16 working set then lives in the 192MB L2).
// ---------------------------------------------------------------------------
__global__ __launch_bounds__(256) void crcl_x_to_bf16(
    const float* __restrict__ X, unsigned short* __restrict__ Xb)
{
    const size_t i = ((size_t)blockIdx.x * 256 + threadIdx.x) * 8;
    float4 a = *(const float4*)(X + i);
    float4 b = *(const float4*)(X + i + 4);
    unsigned short o[8];
    o[0]=f32_to_bf16(a.x); o[1]=f32_to_bf16(a.y);
    o[2]=f32_to_bf16(a.z); o[3]=f32_to_bf16(a.w);
    o[4]=f32_to_bf16(b.x); o[5]=f32_to_bf16(b.y);
    o[6]=f32_to_bf16(b.z); o[7]=f32_to_bf16(b.w);
    *(uint4*)(Xb + i) = *(const uint4*)o;
}

// ---------------------------------------------------------------------------
// Kernel 2 (fast path): Out = Xb @ Wt^T + bias, both operands bf16 in memory.
// 256 threads = 8 wave32; block tile 128x128, K-step 64 -> 16 WMMAs/barrier.
// Wave (2x4 grid) owns 64x32 = 4x2 accumulators.
// ---------------------------------------------------------------------------
#define BK64   64
#define LDSP64 72              // 64 + 8 pad; row stride 144B (16B-aligned, 16 banks)

__global__ __launch_bounds__(256) void crcl_gemm_bf16x(
    const unsigned short* __restrict__ Xb, // (M_TOT, S) bf16
    const unsigned short* __restrict__ Wt, // (S, S) bf16, N-major
    const float* __restrict__ bias,        // (S,)
    float* __restrict__ Out)               // (M_TOT, S) fp32
{
    __shared__ unsigned short Alds[BM * LDSP64];
    __shared__ unsigned short Blds[BN * LDSP64];

    const int tid  = threadIdx.x;
    const int lane = tid & 31;
    const int wave = tid >> 5;
    const int lrow = lane & 15;
    const int lhi  = lane >> 4;

    const int m0 = blockIdx.y * BM;
    const int n0 = blockIdx.x * BN;
    const int wm = (wave >> 2) * 64;
    const int wn = (wave & 3) * 32;

    const v8f vzero = {0.f,0.f,0.f,0.f,0.f,0.f,0.f,0.f};
    v8f acc[4][2];
#pragma unroll
    for (int mt = 0; mt < 4; ++mt)
#pragma unroll
        for (int nt = 0; nt < 2; ++nt) acc[mt][nt] = vzero;

    // Loaders: 2 threads/row, 32 bf16 (64B) each.
    const int ldRow = tid >> 1;
    const int ldCol = (tid & 1) * 32;
    const unsigned short* Xrow = Xb + (size_t)(m0 + ldRow) * S_DIM + ldCol;
    const unsigned short* Wrow = Wt + (size_t)(n0 + ldRow) * S_DIM + ldCol;
    unsigned short* AldsW = &Alds[ldRow * LDSP64 + ldCol];
    unsigned short* BldsW = &Blds[ldRow * LDSP64 + ldCol];

    for (int k0 = 0; k0 < S_DIM; k0 += BK64) {
        if (k0 + BK64 < S_DIM) {
            __builtin_prefetch(Xrow + k0 + BK64, 0, 1);
            __builtin_prefetch(Wrow + k0 + BK64, 0, 1);
        }
        uint4 a0 = *(const uint4*)(Xrow + k0 + 0);
        uint4 a1 = *(const uint4*)(Xrow + k0 + 8);
        uint4 a2 = *(const uint4*)(Xrow + k0 + 16);
        uint4 a3 = *(const uint4*)(Xrow + k0 + 24);
        uint4 b0 = *(const uint4*)(Wrow + k0 + 0);
        uint4 b1 = *(const uint4*)(Wrow + k0 + 8);
        uint4 b2 = *(const uint4*)(Wrow + k0 + 16);
        uint4 b3 = *(const uint4*)(Wrow + k0 + 24);
        *(uint4*)(AldsW + 0)  = a0;  *(uint4*)(AldsW + 8)  = a1;
        *(uint4*)(AldsW + 16) = a2;  *(uint4*)(AldsW + 24) = a3;
        *(uint4*)(BldsW + 0)  = b0;  *(uint4*)(BldsW + 8)  = b1;
        *(uint4*)(BldsW + 16) = b2;  *(uint4*)(BldsW + 24) = b3;

        __syncthreads();

#pragma unroll
        for (int kb = 0; kb < 2; ++kb) {        // two 16x16x32 sub-steps
            FragU afrag[4], bfrag[2];
#pragma unroll
            for (int mt = 0; mt < 4; ++mt) {
                const int row = wm + mt * 16 + lrow;
                const unsigned short* p = &Alds[row * LDSP64 + kb * 32 + lhi * 8];
                afrag[mt].q[0] = *(const uint4*)(p + 0);
                afrag[mt].q[1] = *(const uint4*)(p + 16);
            }
#pragma unroll
            for (int nt = 0; nt < 2; ++nt) {
                const int n = wn + nt * 16 + lrow;
                const unsigned short* p = &Blds[n * LDSP64 + kb * 32 + lhi * 8];
                bfrag[nt].q[0] = *(const uint4*)(p + 0);
                bfrag[nt].q[1] = *(const uint4*)(p + 16);
            }
#pragma unroll
            for (int mt = 0; mt < 4; ++mt)
#pragma unroll
                for (int nt = 0; nt < 2; ++nt)
                    acc[mt][nt] = __builtin_amdgcn_wmma_f32_16x16x32_bf16(
                        false, afrag[mt].v, false, bfrag[nt].v,
                        (short)0, acc[mt][nt], false, false);
        }

        __syncthreads();
    }

#pragma unroll
    for (int nt = 0; nt < 2; ++nt) {
        const int col = n0 + wn + nt * 16 + lrow;
        const float bv = bias[col];
#pragma unroll
        for (int mt = 0; mt < 4; ++mt) {
            const int rbase = m0 + wm + mt * 16 + lhi * 8;
#pragma unroll
            for (int v = 0; v < 8; ++v)
                Out[(size_t)(rbase + v) * S_DIM + col] = acc[mt][nt][v] + bv;
        }
    }
}

// ---------------------------------------------------------------------------
// Kernel 2 (fallback, small ws): inline fp32->bf16 conversion, BK=32.
// ---------------------------------------------------------------------------
#define BK32   32
#define LDSP32 40

__global__ __launch_bounds__(256) void crcl_gemm_bf16(
    const float* __restrict__ X,
    const unsigned short* __restrict__ Wt,
    const float* __restrict__ bias,
    float* __restrict__ Out)
{
    __shared__ unsigned short Alds[BM * LDSP32];
    __shared__ unsigned short Blds[BN * LDSP32];

    const int tid  = threadIdx.x;
    const int lane = tid & 31;
    const int wave = tid >> 5;
    const int lrow = lane & 15;
    const int lhi  = lane >> 4;

    const int m0 = blockIdx.y * BM;
    const int n0 = blockIdx.x * BN;
    const int wm = (wave >> 2) * 64;
    const int wn = (wave & 3) * 32;

    const v8f vzero = {0.f,0.f,0.f,0.f,0.f,0.f,0.f,0.f};
    v8f acc[4][2];
#pragma unroll
    for (int mt = 0; mt < 4; ++mt)
#pragma unroll
        for (int nt = 0; nt < 2; ++nt) acc[mt][nt] = vzero;

    const int ldRow = tid >> 1;
    const int ldCol = (tid & 1) * 16;
    const float*          Xrow = X  + (size_t)(m0 + ldRow) * S_DIM + ldCol;
    const unsigned short* Wrow = Wt + (size_t)(n0 + ldRow) * S_DIM + ldCol;
    unsigned short* AldsW = &Alds[ldRow * LDSP32 + ldCol];
    unsigned short* BldsW = &Blds[ldRow * LDSP32 + ldCol];

    for (int k0 = 0; k0 < S_DIM; k0 += BK32) {
        float4 x0 = *(const float4*)(Xrow + k0 + 0);
        float4 x1 = *(const float4*)(Xrow + k0 + 4);
        float4 x2 = *(const float4*)(Xrow + k0 + 8);
        float4 x3 = *(const float4*)(Xrow + k0 + 12);
        unsigned short ab[16];
        ab[ 0]=f32_to_bf16(x0.x); ab[ 1]=f32_to_bf16(x0.y);
        ab[ 2]=f32_to_bf16(x0.z); ab[ 3]=f32_to_bf16(x0.w);
        ab[ 4]=f32_to_bf16(x1.x); ab[ 5]=f32_to_bf16(x1.y);
        ab[ 6]=f32_to_bf16(x1.z); ab[ 7]=f32_to_bf16(x1.w);
        ab[ 8]=f32_to_bf16(x2.x); ab[ 9]=f32_to_bf16(x2.y);
        ab[10]=f32_to_bf16(x2.z); ab[11]=f32_to_bf16(x2.w);
        ab[12]=f32_to_bf16(x3.x); ab[13]=f32_to_bf16(x3.y);
        ab[14]=f32_to_bf16(x3.z); ab[15]=f32_to_bf16(x3.w);
        *(uint4*)(AldsW + 0) = *(const uint4*)(ab + 0);
        *(uint4*)(AldsW + 8) = *(const uint4*)(ab + 8);
        uint4 b0 = *(const uint4*)(Wrow + k0 + 0);
        uint4 b1 = *(const uint4*)(Wrow + k0 + 8);
        *(uint4*)(BldsW + 0) = b0;
        *(uint4*)(BldsW + 8) = b1;

        __syncthreads();

        FragU afrag[4], bfrag[2];
#pragma unroll
        for (int mt = 0; mt < 4; ++mt) {
            const int row = wm + mt * 16 + lrow;
            const unsigned short* p = &Alds[row * LDSP32 + lhi * 8];
            afrag[mt].q[0] = *(const uint4*)(p + 0);
            afrag[mt].q[1] = *(const uint4*)(p + 16);
        }
#pragma unroll
        for (int nt = 0; nt < 2; ++nt) {
            const int n = wn + nt * 16 + lrow;
            const unsigned short* p = &Blds[n * LDSP32 + lhi * 8];
            bfrag[nt].q[0] = *(const uint4*)(p + 0);
            bfrag[nt].q[1] = *(const uint4*)(p + 16);
        }
#pragma unroll
        for (int mt = 0; mt < 4; ++mt)
#pragma unroll
            for (int nt = 0; nt < 2; ++nt)
                acc[mt][nt] = __builtin_amdgcn_wmma_f32_16x16x32_bf16(
                    false, afrag[mt].v, false, bfrag[nt].v,
                    (short)0, acc[mt][nt], false, false);

        __syncthreads();
    }

#pragma unroll
    for (int nt = 0; nt < 2; ++nt) {
        const int col = n0 + wn + nt * 16 + lrow;
        const float bv = bias[col];
#pragma unroll
        for (int mt = 0; mt < 4; ++mt) {
            const int rbase = m0 + wm + mt * 16 + lhi * 8;
#pragma unroll
            for (int v = 0; v < 8; ++v)
                Out[(size_t)(rbase + v) * S_DIM + col] = acc[mt][nt][v] + bv;
        }
    }
}

// ---------------------------------------------------------------------------
extern "C" void kernel_launch(void* const* d_in, const int* in_sizes, int n_in,
                              void* d_out, int out_size, void* d_ws, size_t ws_size,
                              hipStream_t stream) {
    const float* x      = (const float*)d_in[0];   // (8, 1024, 2048)
    const float* weight = (const float*)d_in[1];   // (2, 2048)
    const float* bias   = (const float*)d_in[2];   // (2048,)
    const float* decay  = (const float*)d_in[3];   // (2, 1)

    unsigned short* wt = (unsigned short*)d_ws;    // 8 MB: (S,S) bf16
    float* out = (float*)d_out;

    const size_t wt_bytes = (size_t)S_DIM * S_DIM * 2;          // 8 MB
    const size_t xb_bytes = (size_t)M_TOT * S_DIM * 2;          // 32 MB

    crcl_build_wt<<<dim3(S_DIM), dim3(256), 0, stream>>>(weight, decay, wt);

    dim3 grid(S_DIM / BN, M_TOT / BM);             // (16, 64)
    if (ws_size >= wt_bytes + xb_bytes) {
        // Fast path: pre-convert X to bf16 once, lean BK=64 GEMM.
        unsigned short* xb = wt + (size_t)S_DIM * S_DIM;
        crcl_x_to_bf16<<<dim3((M_TOT * (S_DIM / 8)) / 256), dim3(256), 0, stream>>>(x, xb);
        crcl_gemm_bf16x<<<grid, dim3(256), 0, stream>>>(xb, wt, bias, out);
    } else {
        // Fallback: convert X tiles inline (BK=32).
        crcl_gemm_bf16<<<grid, dim3(256), 0, stream>>>(x, wt, bias, out);
    }
}
